// QuantCNV_65300682768583
// MI455X (gfx1250) — 8-bit matrix operand order repaired and checked by reading, compiled, not run on hardware
//
#include <hip/hip_runtime.h>
#include <cstdint>
#include <cstddef>

typedef __attribute__((ext_vector_type(8))) int v8i;
typedef __attribute__((ext_vector_type(4))) int v4i_t;
typedef __attribute__((ext_vector_type(4))) unsigned int v4u_t;

#define EPS_BN 1e-4f

__global__ void __launch_bounds__(256)
k_maxabs(const float* __restrict__ w, int count, float* __restrict__ outScale) {
  __shared__ float sm[256];
  float v = 0.f;
  for (int i = threadIdx.x; i < count; i += 256) v = fmaxf(v, fabsf(w[i]));
  sm[threadIdx.x] = v;
  __syncthreads();
  for (int s = 128; s > 0; s >>= 1) {
    if ((int)threadIdx.x < s) sm[threadIdx.x] = fmaxf(sm[threadIdx.x], sm[threadIdx.x + s]);
    __syncthreads();
  }
  if (threadIdx.x == 0) outScale[0] = sm[0];
}

__global__ void __launch_bounds__(256)
k_pack(const float* __restrict__ w, const float* __restrict__ wscale,
       int8_t* __restrict__ wpack, int Co, int Ci, int FH, int CoTiles, int KC) {
  const int taps = FH * FH;
  const int total = CoTiles * taps * KC * 256;
  const int tid = blockIdx.x * 256 + threadIdx.x;
  if (tid >= total) return;
  const int j    = tid & 7;
  const int lane = (tid >> 3) & 31;
  int rest = tid >> 8;
  const int kc  = rest % KC;  rest /= KC;
  const int tap = rest % taps;
  const int ct  = rest / taps;
  const int dy = tap / FH, dx = tap - dy * FH;
  const int co = ct * 16 + (lane & 15);
  const int kbase = ((lane >> 4) << 3) + ((j >> 1) << 4) + ((j & 1) << 2);
  const float s = wscale[0];
  const float inv = (s > 0.f) ? (1.0f / s) : 0.f;
  unsigned dw = 0;
  for (int t = 0; t < 4; ++t) {
    const int k = kc * 64 + kbase + t;
    int v = 0;
    if (co < Co && k < Ci) {
      const float wv = w[(((size_t)co * Ci + k) * FH + dy) * FH + dx];
      float q = rintf(wv * inv);
      q = fminf(1.f, fmaxf(-1.f, q));
      v = (int)q;
    }
    dw |= ((unsigned)(v & 0xFF)) << (8 * t);
  }
  ((unsigned*)wpack)[tid] = dw;
}

__global__ void __launch_bounds__(256)
k_qin(const float* __restrict__ x, int8_t* __restrict__ q) {
  const int pix = blockIdx.x * 256 + threadIdx.x;
  if (pix >= 1024 * 32 * 32) return;
  const int n  = pix >> 10;
  const int yx = pix & 1023;
  const float inv_s = 127.0f / (1.0f - 0.0078125f);
  unsigned dw0 = 0;
  for (int ch = 0; ch < 3; ++ch) {
    const float xv = x[((size_t)n * 3 + ch) * 1024 + yx];
    float qv = rintf(xv * inv_s);
    qv = fminf(127.f, fmaxf(-127.f, qv));
    dw0 |= ((unsigned)(((int)qv) & 0xFF)) << (8 * ch);
  }
  unsigned* o = (unsigned*)(q + (size_t)pix * 64);
  o[0] = dw0;
#pragma unroll
  for (int i = 1; i < 16; ++i) o[i] = 0;
}

__global__ void __launch_bounds__(256)
k_pool(const int8_t* __restrict__ in, int8_t* __restrict__ out, int H, int W, int C) {
  const int Ho = H >> 1, Wo = W >> 1;
  const size_t total = (size_t)1024 * Ho * Wo * C;
  const size_t tid = (size_t)blockIdx.x * 256 + threadIdx.x;
  if (tid >= total) return;
  const int c = (int)(tid % C);
  size_t r = tid / C;
  const int xo = (int)(r % Wo); r /= Wo;
  const int yo = (int)(r % Ho);
  const int n  = (int)(r / Ho);
  const int8_t* base = in + ((((size_t)n * H) + yo * 2) * W + xo * 2) * C + c;
  const int v0 = base[0], v1 = base[C];
  const int v2 = base[(size_t)W * C], v3 = base[(size_t)W * C + C];
  int m01 = v0 > v1 ? v0 : v1;
  int m23 = v2 > v3 ? v2 : v3;
  out[tid] = (int8_t)(m01 > m23 ? m01 : m23);
}

__global__ void __launch_bounds__(256)
k_conv_wmma(const int8_t* __restrict__ act, const int8_t* __restrict__ wpack,
            int8_t* __restrict__ out8, float* __restrict__ outf,
            const float* __restrict__ wscale,
            const float* __restrict__ g, const float* __restrict__ bta,
            const float* __restrict__ mu, const float* __restrict__ var,
            float sa, int Hin, int Win, int Cin, int CoPad, int CoReal,
            int FH, int mode) {
  extern __shared__ int8_t lds[];
  const int lane = threadIdx.x & 31;
  const int wave = threadIdx.x >> 5;
  const int Hout = Hin - FH + 1;
  const int Wout = Win - FH + 1;
  const int HWo  = Hout * Wout;
  const int P    = 1024 * HWo;
  const int mTiles = (P + 15) >> 4;
  const int coTile = blockIdx.y;
  const int KC   = Cin >> 6;
  const int taps = FH * FH;
  const int stageBytes = taps * KC * 1024;
  const int8_t* wsrc = wpack + (size_t)coTile * stageBytes;

  if (wave == 0) {
    const unsigned ldsAddr = (unsigned)(uintptr_t)(&lds[0]);
    const unsigned long long ga = (unsigned long long)(uintptr_t)wsrc;
    const unsigned ne8 = (unsigned)(stageBytes >> 3);
    v4u_t g0;
    g0.x = 1u;
    g0.y = ldsAddr;
    g0.z = (unsigned)(ga & 0xFFFFFFFFu);
    g0.w = (unsigned)((ga >> 32) & 0x01FFFFFFu) | (2u << 30);
    v8i g1 = {};
    g1[0] = (3 << 16);
    g1[1] = (int)((ne8 & 0xFFFFu) << 16);
    g1[2] = (int)(((ne8 >> 16) & 0xFFFFu) | (1u << 16));
    g1[3] = (int)((ne8 & 0xFFFFu) << 16);
    g1[4] = 1;
    g1[5] = (int)ne8;
    v4i_t gz = {};
#if __has_include(<hip/amd_detail/amd_gfx1250_TDM.h>)
    v8i gz8 = {};
    __builtin_amdgcn_tensor_load_to_lds(g0, g1, gz, gz, gz8, 0);
#else
    __builtin_amdgcn_tensor_load_to_lds(g0, g1, gz, gz, 0);
#endif
    __builtin_amdgcn_s_wait_tensorcnt(0);
  }
  __syncthreads();
  {
    const int nInt4 = stageBytes >> 4;
    for (int i = threadIdx.x; i < nInt4; i += 256)
      ((int4*)lds)[i] = ((const int4*)wsrc)[i];
  }
  __syncthreads();

  const int mt0 = (blockIdx.x * 8 + wave) * 2;
  const int mt1 = mt0 + 1;
  const int mt0c = (mt0 < mTiles) ? mt0 : (mTiles - 1);
  const int mt1c = (mt1 < mTiles) ? mt1 : (mTiles - 1);

  const int klo = (lane >> 4) << 3;
  int p0 = mt0c * 16 + (lane & 15);
  p0 = (p0 < P) ? p0 : (P - 1);
  const int n0  = p0 / HWo;
  const int r0  = p0 - n0 * HWo;
  const int oy0 = r0 / Wout;
  const int ox0 = r0 - oy0 * Wout;
  int p1 = mt1c * 16 + (lane & 15);
  p1 = (p1 < P) ? p1 : (P - 1);
  const int n1  = p1 / HWo;
  const int r1  = p1 - n1 * HWo;
  const int oy1 = r1 / Wout;
  const int ox1 = r1 - oy1 * Wout;

  v8i acc0 = {}, acc1 = {};
  for (int tap = 0; tap < taps; ++tap) {
    const int dy = tap / FH, dx = tap - dy * FH;
    const int8_t* rowp0 =
        act + ((((size_t)n0 * Hin) + (oy0 + dy)) * Win + (ox0 + dx)) * Cin + klo;
    const int8_t* rowp1 =
        act + ((((size_t)n1 * Hin) + (oy1 + dy)) * Win + (ox1 + dx)) * Cin + klo;
    const int8_t* bptr = lds + (size_t)tap * KC * 1024 + (size_t)lane * 32;
    for (int kc = 0; kc < KC; ++kc) {
      const int8_t* ap0 = rowp0 + kc * 64;
      const int8_t* ap1 = rowp1 + kc * 64;
      __builtin_prefetch(ap0 + 256, 0, 0);
      const uint2 a00 = *(const uint2*)(ap0 +  0);
      const uint2 a01 = *(const uint2*)(ap0 + 16);
      const uint2 a02 = *(const uint2*)(ap0 + 32);
      const uint2 a03 = *(const uint2*)(ap0 + 48);
      const uint2 a10 = *(const uint2*)(ap1 +  0);
      const uint2 a11 = *(const uint2*)(ap1 + 16);
      const uint2 a12 = *(const uint2*)(ap1 + 32);
      const uint2 a13 = *(const uint2*)(ap1 + 48);
      const int4 w0 = ((const int4*)bptr)[0];
      const int4 w1 = ((const int4*)bptr)[1];
      v8i a, b;
      b[0] = w0.x; b[1] = w0.y; b[2] = w0.z; b[3] = w0.w;
      b[4] = w1.x; b[5] = w1.y; b[6] = w1.z; b[7] = w1.w;
      a[0] = (int)a00.x; a[1] = (int)a00.y; a[2] = (int)a01.x; a[3] = (int)a01.y;
      a[4] = (int)a02.x; a[5] = (int)a02.y; a[6] = (int)a03.x; a[7] = (int)a03.y;
      acc0 = __builtin_amdgcn_wmma_i32_16x16x64_iu8(true, a, true, b, acc0, false, false);
      a[0] = (int)a10.x; a[1] = (int)a10.y; a[2] = (int)a11.x; a[3] = (int)a11.y;
      a[4] = (int)a12.x; a[5] = (int)a12.y; a[6] = (int)a13.x; a[7] = (int)a13.y;
      acc1 = __builtin_amdgcn_wmma_i32_16x16x64_iu8(true, a, true, b, acc1, false, false);
      bptr += 1024;
    }
  }

  const float comb = sa * wscale[0];
  const int c    = coTile * 16 + (lane & 15);
  const int mOff = (lane >> 4) << 3;
  const int bi   = mode ? 0 : c;
  const float sbn = g[bi] * rsqrtf(var[bi] + EPS_BN);
  const float mm  = mu[bi];
  const float bb  = bta[bi];
#pragma unroll
  for (int r = 0; r < 8; ++r) {
    for (int t = 0; t < 2; ++t) {
      const int pm = (t ? mt1 : mt0) * 16 + mOff + r;
      if (pm >= P || c >= CoReal) continue;
      float h = (float)(t ? acc1[r] : acc0[r]) * comb;
      h = (h - mm) * sbn + bb;
      if (mode == 0) {
        float q = rintf(h);
        q = fminf(1.f, fmaxf(-1.f, q));
        out8[(size_t)pm * CoPad + c] = (int8_t)(int)q;
      } else {
        outf[(size_t)pm * CoReal + c] = h;
      }
    }
  }
}

extern "C" void kernel_launch(void* const* d_in, const int* in_sizes, int n_in,
                              void* d_out, int out_size, void* d_ws, size_t ws_size,
                              hipStream_t stream) {
  (void)in_sizes; (void)n_in; (void)out_size; (void)ws_size;

  const float* x = (const float*)d_in[0];
  const float *cw[6], *bg[6], *bb[6], *bm[6], *bv[6];
  for (int li = 0; li < 6; ++li) {
    cw[li] = (const float*)d_in[1 + li * 5 + 0];
    bg[li] = (const float*)d_in[1 + li * 5 + 1];
    bb[li] = (const float*)d_in[1 + li * 5 + 2];
    bm[li] = (const float*)d_in[1 + li * 5 + 3];
    bv[li] = (const float*)d_in[1 + li * 5 + 4];
  }
  const float* fw[3] = {(const float*)d_in[31], (const float*)d_in[32],
                        (const float*)d_in[33]};
  const float *fg[2], *fb[2], *fm[2], *fv[2];
  for (int li = 0; li < 2; ++li) {
    fg[li] = (const float*)d_in[34 + li * 4 + 0];
    fb[li] = (const float*)d_in[34 + li * 4 + 1];
    fm[li] = (const float*)d_in[34 + li * 4 + 2];
    fv[li] = (const float*)d_in[34 + li * 4 + 3];
  }
  const float* tnw = (const float*)d_in[42];
  const float* tnb = (const float*)d_in[43];
  const float* tnm = (const float*)d_in[44];
  const float* tnv = (const float*)d_in[45];

  char*   ws = (char*)d_ws;
  float*  sc = (float*)ws;
  int8_t* wp = (int8_t*)(ws + 256);
  static const size_t wsz[9] = {36864, 36864, 73728, 147456, 294912,
                                589824, 131072, 262144, 8192};
  size_t wpo[9]; { size_t o = 0; for (int i = 0; i < 9; ++i) { wpo[i] = o; o += wsz[i]; } }
  int8_t* bufA = (int8_t*)(ws + ((size_t)4u << 20));
  int8_t* bufB = bufA + ((size_t)72u << 20);

  const int Cos[6] = {64, 64, 128, 128, 256, 256};
  const int Cis[6] = {3, 64, 64, 128, 128, 256};
  for (int li = 0; li < 6; ++li)
    k_maxabs<<<1, 256, 0, stream>>>(cw[li], Cos[li] * Cis[li] * 9, sc + li);
  k_maxabs<<<1, 256, 0, stream>>>(fw[0], 512 * 256, sc + 6);
  k_maxabs<<<1, 256, 0, stream>>>(fw[1], 512 * 512, sc + 7);
  k_maxabs<<<1, 256, 0, stream>>>(fw[2], 10 * 512, sc + 8);

  auto packL = [&](const float* w, const float* s, size_t wi, int Co, int Ci,
                   int FH, int CT, int KC) {
    int total = CT * FH * FH * KC * 256;
    k_pack<<<(total + 255) / 256, 256, 0, stream>>>(w, s, wp + wpo[wi], Co, Ci, FH, CT, KC);
  };
  packL(cw[0], sc + 0, 0,  64,   3, 3,  4, 1);
  packL(cw[1], sc + 1, 1,  64,  64, 3,  4, 1);
  packL(cw[2], sc + 2, 2, 128,  64, 3,  8, 1);
  packL(cw[3], sc + 3, 3, 128, 128, 3,  8, 2);
  packL(cw[4], sc + 4, 4, 256, 128, 3, 16, 2);
  packL(cw[5], sc + 5, 5, 256, 256, 3, 16, 4);
  packL(fw[0], sc + 6, 6, 512, 256, 1, 32, 4);
  packL(fw[1], sc + 7, 7, 512, 512, 1, 32, 8);
  packL(fw[2], sc + 8, 8,  10, 512, 1,  1, 8);

  auto convL = [&](const int8_t* a, size_t wi, int8_t* o8, float* of,
                   const float* s, const float* g, const float* b2,
                   const float* m2, const float* v2, float sa,
                   int Hin, int Win, int Cin, int CoPad, int CoReal, int FH, int mode) {
    const int Hout = Hin - FH + 1, Wout = Win - FH + 1;
    const int P = 1024 * Hout * Wout;
    const int mTiles = (P + 15) / 16;
    const int shmem = FH * FH * (Cin >> 6) * 1024;
    dim3 grid((mTiles + 15) / 16, CoPad / 16);
    k_conv_wmma<<<grid, 256, shmem, stream>>>(a, wp + wpo[wi], o8, of, s, g, b2, m2, v2,
                                              sa, Hin, Win, Cin, CoPad, CoReal, FH, mode);
  };
  auto poolL = [&](const int8_t* in, int8_t* out, int H, int W, int C) {
    size_t total = (size_t)1024 * (H / 2) * (W / 2) * C;
    k_pool<<<(unsigned)((total + 255) / 256), 256, 0, stream>>>(in, out, H, W, C);
  };

  const float sa0 = (1.0f - 0.0078125f) / 127.0f;

  k_qin<<<4096, 256, 0, stream>>>(x, bufA);
  convL(bufA, 0, bufB, nullptr, sc + 0, bg[0], bb[0], bm[0], bv[0], sa0,
        32, 32, 64, 64, 64, 3, 0);
  convL(bufB, 1, bufA, nullptr, sc + 1, bg[1], bb[1], bm[1], bv[1], 1.f,
        30, 30, 64, 64, 64, 3, 0);
  poolL(bufA, bufB, 28, 28, 64);
  convL(bufB, 2, bufA, nullptr, sc + 2, bg[2], bb[2], bm[2], bv[2], 1.f,
        14, 14, 64, 128, 128, 3, 0);
  convL(bufA, 3, bufB, nullptr, sc + 3, bg[3], bb[3], bm[3], bv[3], 1.f,
        12, 12, 128, 128, 128, 3, 0);
  poolL(bufB, bufA, 10, 10, 128);
  convL(bufA, 4, bufB, nullptr, sc + 4, bg[4], bb[4], bm[4], bv[4], 1.f,
        5, 5, 128, 256, 256, 3, 0);
  convL(bufB, 5, bufA, nullptr, sc + 5, bg[5], bb[5], bm[5], bv[5], 1.f,
        3, 3, 256, 256, 256, 3, 0);
  convL(bufA, 6, bufB, nullptr, sc + 6, fg[0], fb[0], fm[0], fv[0], 1.f,
        1, 1, 256, 512, 512, 1, 0);
  convL(bufB, 7, bufA, nullptr, sc + 7, fg[1], fb[1], fm[1], fv[1], 1.f,
        1, 1, 512, 512, 512, 1, 0);
  convL(bufA, 8, nullptr, (float*)d_out, sc + 8, tnw, tnb, tnm, tnv, 1.f,
        1, 1, 512, 16, 10, 1, 1);
}
